// TransMILCox_71932112273690
// MI455X (gfx1250) — matrix operand order repaired and checked by reading, not compiled, not run on hardware
//
#include <hip/hip_runtime.h>
#include <math.h>

// Problem constants (fixed by the reference)
#define DD   768      // model dim
#define SS   2048     // input seq
#define GH   46       // grid H = ceil(sqrt(2048))
#define GW   45       // grid W
#define SP   2070     // H*W feature tokens
#define NN   2071     // tokens incl cls
#define NP   2080     // padded attention width (mult of 32)
#define NR   2128     // padded row count: ceil(2071/64)*64 + 16 (cls row offset slack)
#define QKD  256      // sparse-attn qk dim
#define TD   2304     // 3*D
#define FFD  512
#define HDIM 48       // head dim
#define NH   16       // heads

typedef __attribute__((ext_vector_type(16))) __bf16 v16bf;
typedef __attribute__((ext_vector_type(8)))  float  v8f;

// ---------------------------------------------------------------------------
// Fragment loaders: fp32 memory -> bf16 WMMA fragments (hardware cvt via cast).
// A (16x32, MxK): lane half 0 -> K{0..7,16..23}, half 1 -> K{8..15,24..31}
// B (32x16, KxN): lane half 0 -> K{0..7,16..23}, half 1 -> K{8..15,24..31}
// ---------------------------------------------------------------------------
__device__ __forceinline__ v16bf load_a_frag(const float* __restrict__ arow, int k0, int hl)
{
    const float4* p  = (const float4*)(arow + k0 + hl * 8);
    const float4* p2 = (const float4*)(arow + k0 + 16 + hl * 8);
    float4 t0 = p[0], t1 = p[1], t2 = p2[0], t3 = p2[1];
    float f[16] = {t0.x,t0.y,t0.z,t0.w, t1.x,t1.y,t1.z,t1.w,
                   t2.x,t2.y,t2.z,t2.w, t3.x,t3.y,t3.z,t3.w};
    v16bf r;
#pragma unroll
    for (int i = 0; i < 16; i++) r[i] = (__bf16)f[i];
    return r;
}

__device__ __forceinline__ v16bf load_b_frag(const float* __restrict__ wrow, int k0, int hl)
{
    const float4* q  = (const float4*)(wrow + k0 + hl * 8);
    const float4* q2 = (const float4*)(wrow + k0 + 16 + hl * 8);
    float4 u0 = q[0], u1 = q[1], u2 = q2[0], u3 = q2[1];
    float f[16] = {u0.x,u0.y,u0.z,u0.w, u1.x,u1.y,u1.z,u1.w,
                   u2.x,u2.y,u2.z,u2.w, u3.x,u3.y,u3.z,u3.w};
    v16bf r;
#pragma unroll
    for (int i = 0; i < 16; i++) r[i] = (__bf16)f[i];
    return r;
}

__device__ __forceinline__ v16bf load_a_frag_tail(const float* __restrict__ arow,
                                                  int k0, int hl, int K)
{
    v16bf r;
#pragma unroll
    for (int e = 0; e < 8; e++) {
        int ka = k0 + hl * 8 + e;
        int kb = k0 + 16 + hl * 8 + e;
        r[e]     = (ka < K) ? (__bf16)arow[ka] : (__bf16)0.f;
        r[8 + e] = (kb < K) ? (__bf16)arow[kb] : (__bf16)0.f;
    }
    return r;
}

__device__ __forceinline__ v16bf load_b_frag_tail(const float* __restrict__ wrow,
                                                  int k0, int hl, int K)
{
    v16bf r;
#pragma unroll
    for (int e = 0; e < 8; e++) {
        int ka = k0 + hl * 8 + e;
        int kb = k0 + 16 + hl * 8 + e;
        r[e]     = (ka < K) ? (__bf16)wrow[ka] : (__bf16)0.f;
        r[8 + e] = (kb < K) ? (__bf16)wrow[kb] : (__bf16)0.f;
    }
    return r;
}

#define WMMA_BF16(A_, B_, C_) \
    __builtin_amdgcn_wmma_f32_16x16x32_bf16(false, (A_), false, (B_), (short)0, (C_), false, false)

// ---------------------------------------------------------------------------
// WMMA GEMM:  C[M,Nc] = act(alpha * A[M,K] @ W[Nc,K]^T + bias) (+= C if accum)
// Block = 128 threads = 4 waves arranged 2x2; each wave computes a 32x32 tile
// (2 A-frags x 2 B-frags -> 4 WMMAs per 32-wide K step). Block tile = 64x64.
// Requirements: Nc multiple of 16 (weights are multiples of 32); A buffers
// row-padded to the 64-aligned tile edge; 16B-aligned row strides; K arbitrary.
// ---------------------------------------------------------------------------
__global__ __launch_bounds__(128)
void gemm_bf16_wmma(const float* __restrict__ A, int lda,
                    const float* __restrict__ Wm, int ldb,
                    const float* __restrict__ bias,
                    float* __restrict__ C, int ldc,
                    int M, int Nc, int K,
                    float alpha, int act, int accum)
{
    const int lane = threadIdx.x & 31;
    const int wave = threadIdx.x >> 5;
    const int wx = wave & 1, wy = wave >> 1;
    const int n0 = blockIdx.x * 64 + wx * 32;
    if (n0 >= Nc) return;
    const int m0  = blockIdx.y * 64 + wy * 32;
    const int hl  = lane >> 4;
    const int l15 = lane & 15;

    const float* __restrict__ a0row = A  + (size_t)(m0 + l15) * lda;
    const float* __restrict__ a1row = A  + (size_t)(m0 + 16 + l15) * lda;
    const float* __restrict__ b0row = Wm + (size_t)(n0 + l15) * ldb;
    const float* __restrict__ b1row = Wm + (size_t)(n0 + 16 + l15) * ldb;

    v8f acc00 = {0.f,0.f,0.f,0.f,0.f,0.f,0.f,0.f};
    v8f acc01 = acc00, acc10 = acc00, acc11 = acc00;

    int k0 = 0;
    for (; k0 + 32 <= K; k0 += 32) {
        v16bf a0 = load_a_frag(a0row, k0, hl);
        v16bf a1 = load_a_frag(a1row, k0, hl);
        v16bf b0 = load_b_frag(b0row, k0, hl);
        v16bf b1 = load_b_frag(b1row, k0, hl);
        if (k0 + 64 <= K) {
            __builtin_prefetch(a0row + k0 + 32, 0, 1);
            __builtin_prefetch(b0row + k0 + 32, 0, 1);
        }
        acc00 = WMMA_BF16(a0, b0, acc00);
        acc01 = WMMA_BF16(a0, b1, acc01);
        acc10 = WMMA_BF16(a1, b0, acc10);
        acc11 = WMMA_BF16(a1, b1, acc11);
    }
    if (k0 < K) {   // masked tail (e.g. K=48 head dim)
        v16bf a0 = load_a_frag_tail(a0row, k0, hl, K);
        v16bf a1 = load_a_frag_tail(a1row, k0, hl, K);
        v16bf b0 = load_b_frag_tail(b0row, k0, hl, K);
        v16bf b1 = load_b_frag_tail(b1row, k0, hl, K);
        acc00 = WMMA_BF16(a0, b0, acc00);
        acc01 = WMMA_BF16(a0, b1, acc01);
        acc10 = WMMA_BF16(a1, b0, acc10);
        acc11 = WMMA_BF16(a1, b1, acc11);
    }

    // C/D layout: VGPR r -> row (subtile_m + r + 8*hl), col (subtile_n + l15)
#pragma unroll
    for (int i = 0; i < 2; i++) {
        v8f accb0 = i ? acc10 : acc00;
        v8f accb1 = i ? acc11 : acc01;
        int mi = m0 + i * 16;
#pragma unroll
        for (int j = 0; j < 2; j++) {
            int nj = n0 + j * 16;
            if (nj >= Nc) continue;            // wave-uniform guard (Nc mult of 16)
            v8f acc = j ? accb1 : accb0;
            int n = nj + l15;
#pragma unroll
            for (int r = 0; r < 8; r++) {
                int m = mi + r + hl * 8;
                if (m < M) {
                    float v = alpha * acc[r];
                    if (bias) v += bias[n];
                    if (act == 1) v = 0.5f * v * (1.0f + erff(v * 0.70710678118654752f));
                    float* cp = C + (size_t)m * ldc + n;
                    if (accum) v += *cp;
                    *cp = v;
                }
            }
        }
    }
}

// ---------------------------------------------------------------------------
// Positional encoding: pos = f + dwconv7 + dwconv5 + dwconv3 on 46x45 grid.
// Writes rows 1..SP of out (row 0 = cls, written separately).
// ---------------------------------------------------------------------------
__global__ void posenc_kernel(const float* __restrict__ xin,
                              const float* __restrict__ w7, const float* __restrict__ b7,
                              const float* __restrict__ w5, const float* __restrict__ b5,
                              const float* __restrict__ w3, const float* __restrict__ b3,
                              float* __restrict__ out)
{
    int idx = blockIdx.x * blockDim.x + threadIdx.x;
    if (idx >= SP * DD) return;
    int c = idx % DD;
    int p = idx / DD;
    int y = p / GW, x = p % GW;
    auto g = [&](int yy, int xx) -> float {
        if (yy < 0 || yy >= GH || xx < 0 || xx >= GW) return 0.f;
        int s = yy * GW + xx;
        return (s < SS) ? xin[(size_t)s * DD + c] : 0.f;   // padded tokens are zero
    };
    float base = g(y, x);
    float a7 = 0.f;
    for (int ky = 0; ky < 7; ky++)
        for (int kx = 0; kx < 7; kx++)
            a7 += g(y + ky - 3, x + kx - 3) * w7[c * 49 + ky * 7 + kx];
    float a5 = 0.f;
    for (int ky = 0; ky < 5; ky++)
        for (int kx = 0; kx < 5; kx++)
            a5 += g(y + ky - 2, x + kx - 2) * w5[c * 25 + ky * 5 + kx];
    float a3 = 0.f;
    for (int ky = 0; ky < 3; ky++)
        for (int kx = 0; kx < 3; kx++)
            a3 += g(y + ky - 1, x + kx - 1) * w3[c * 9 + ky * 3 + kx];
    out[(size_t)(1 + p) * DD + c] = base + a7 + b7[c] + a5 + b5[c] + a3 + b3[c];
}

// LayerNorm over DD, optional elementwise residual R added first.
__global__ __launch_bounds__(256)
void ln_kernel(const float* __restrict__ X, const float* __restrict__ R,
               const float* __restrict__ g, const float* __restrict__ b,
               float* __restrict__ out, int ld_in, int ld_r, int ld_out)
{
    __shared__ float red[256];
    const int row = blockIdx.x, tid = threadIdx.x;
    const float* xr = X + (size_t)row * ld_in;
    const float* rr = R ? (R + (size_t)row * ld_r) : nullptr;
    float s = 0.f;
    for (int c = tid; c < DD; c += 256) s += xr[c] + (rr ? rr[c] : 0.f);
    red[tid] = s; __syncthreads();
    for (int st = 128; st > 0; st >>= 1) { if (tid < st) red[tid] += red[tid + st]; __syncthreads(); }
    float mean = red[0] / DD; __syncthreads();
    s = 0.f;
    for (int c = tid; c < DD; c += 256) {
        float v = xr[c] + (rr ? rr[c] : 0.f) - mean; s += v * v;
    }
    red[tid] = s; __syncthreads();
    for (int st = 128; st > 0; st >>= 1) { if (tid < st) red[tid] += red[tid + st]; __syncthreads(); }
    float rs = rsqrtf(red[0] / DD + 1e-5f); __syncthreads();
    float* orow = out + (size_t)row * ld_out;
    for (int c = tid; c < DD; c += 256)
        orow[c] = (xr[c] + (rr ? rr[c] : 0.f) - mean) * rs * g[c] + b[c];
}

// Row softmax over ncols (in-place, ld = NP); pad columns written as 0.
// If topk_ptr != null: keep only values >= bisected top-k threshold (sparse softmax).
__global__ __launch_bounds__(256)
void softmax_topk_kernel(float* __restrict__ Smat, int ld, int ncols,
                         const int* __restrict__ topk_ptr)
{
    __shared__ float srow[NP];
    __shared__ float red[256];
    __shared__ int   redi[256];
    const int row = blockIdx.x, tid = threadIdx.x;
    float* rp = Smat + (size_t)row * ld;
    float mx = -3.4e38f, mn = 3.4e38f;
    for (int c = tid; c < NP; c += 256) {
        float v = (c < ncols) ? rp[c] : -3.4e38f;
        srow[c] = v;
        if (c < ncols) { mx = fmaxf(mx, v); mn = fminf(mn, v); }
    }
    red[tid] = mx; __syncthreads();
    for (int st = 128; st > 0; st >>= 1) { if (tid < st) red[tid] = fmaxf(red[tid], red[tid + st]); __syncthreads(); }
    float rmax = red[0]; __syncthreads();
    red[tid] = mn; __syncthreads();
    for (int st = 128; st > 0; st >>= 1) { if (tid < st) red[tid] = fminf(red[tid], red[tid + st]); __syncthreads(); }
    float rmin = red[0]; __syncthreads();

    float thr = -3.4e38f;
    if (topk_ptr) {
        int ksel = topk_ptr[0];
        if (ksel > ncols) ksel = ncols;
        if (ksel < ncols) {
            float lo = rmin, hi = rmax;
            for (int it = 0; it < 32; it++) {
                float mid = 0.5f * (lo + hi);
                int cnt = 0;
                for (int c = tid; c < ncols; c += 256) if (srow[c] >= mid) cnt++;
                redi[tid] = cnt; __syncthreads();
                for (int st = 128; st > 0; st >>= 1) { if (tid < st) redi[tid] += redi[tid + st]; __syncthreads(); }
                int total = redi[0]; __syncthreads();
                if (total >= ksel) lo = mid; else hi = mid;
            }
            thr = lo;
        }
    }
    float s = 0.f;
    for (int c = tid; c < ncols; c += 256) {
        float v = srow[c];
        if (v >= thr) s += expf(v - rmax);
    }
    red[tid] = s; __syncthreads();
    for (int st = 128; st > 0; st >>= 1) { if (tid < st) red[tid] += red[tid + st]; __syncthreads(); }
    float inv = 1.0f / red[0]; __syncthreads();
    for (int c = tid; c < NP; c += 256) {
        float v = srow[c];
        rp[c] = (c < ncols && v >= thr) ? expf(v - rmax) * inv : 0.f;
    }
}

// out[c*ldout + sp] = (sp < rows) ? in[sp*ldin + c] : 0   (zero-fills K padding)
__global__ void transpose_kernel(const float* __restrict__ in, int ldin, int rows,
                                 float* __restrict__ out, int ldout)
{
    int sp = blockIdx.x * blockDim.x + threadIdx.x;
    int c  = blockIdx.y;
    if (sp >= ldout) return;
    out[(size_t)c * ldout + sp] = (sp < rows) ? in[(size_t)sp * ldin + c] : 0.f;
}

__global__ void copy_kernel(const float* __restrict__ in, float* __restrict__ out, int n)
{
    int i = blockIdx.x * blockDim.x + threadIdx.x;
    if (i < n) out[i] = in[i];
}

// Final head: out = LN(x_cls) . head_w
__global__ __launch_bounds__(256)
void head_kernel(const float* __restrict__ x0, const float* __restrict__ g,
                 const float* __restrict__ b, const float* __restrict__ hw,
                 float* __restrict__ outp)
{
    __shared__ float red[256];
    const int tid = threadIdx.x;
    float s = 0.f;
    for (int c = tid; c < DD; c += 256) s += x0[c];
    red[tid] = s; __syncthreads();
    for (int st = 128; st > 0; st >>= 1) { if (tid < st) red[tid] += red[tid + st]; __syncthreads(); }
    float mean = red[0] / DD; __syncthreads();
    s = 0.f;
    for (int c = tid; c < DD; c += 256) { float v = x0[c] - mean; s += v * v; }
    red[tid] = s; __syncthreads();
    for (int st = 128; st > 0; st >>= 1) { if (tid < st) red[tid] += red[tid + st]; __syncthreads(); }
    float rs = rsqrtf(red[0] / DD + 1e-5f); __syncthreads();
    s = 0.f;
    for (int c = tid; c < DD; c += 256)
        s += ((x0[c] - mean) * rs * g[c] + b[c]) * hw[c];
    red[tid] = s; __syncthreads();
    for (int st = 128; st > 0; st >>= 1) { if (tid < st) red[tid] += red[tid + st]; __syncthreads(); }
    if (tid == 0) *outp = red[0];
}

// ---------------------------------------------------------------------------
extern "C" void kernel_launch(void* const* d_in, const int* in_sizes, int n_in,
                              void* d_out, int out_size, void* d_ws, size_t ws_size,
                              hipStream_t stream)
{
    const float* x_all = (const float*)d_in[0];
    const float* cls   = (const float*)d_in[1];
    const float* k7w = (const float*)d_in[2];  const float* k7b = (const float*)d_in[3];
    const float* k5w = (const float*)d_in[4];  const float* k5b = (const float*)d_in[5];
    const float* k3w = (const float*)d_in[6];  const float* k3b = (const float*)d_in[7];
    const float* pre_g = (const float*)d_in[8];  const float* pre_b = (const float*)d_in[9];
    const float* q_w = (const float*)d_in[10]; const float* q_b = (const float*)d_in[11];
    const float* k_w = (const float*)d_in[12]; const float* k_b = (const float*)d_in[13];
    const float* corr_g = (const float*)d_in[14]; const float* corr_b = (const float*)d_in[15];
    const float* qkv_w = (const float*)d_in[16]; const float* qkv_b = (const float*)d_in[17];
    const float* out_w = (const float*)d_in[18]; const float* out_b = (const float*)d_in[19];
    const float* n1_g = (const float*)d_in[20]; const float* n1_b = (const float*)d_in[21];
    const float* n2_g = (const float*)d_in[22]; const float* n2_b = (const float*)d_in[23];
    const float* l1_w = (const float*)d_in[24]; const float* l1_b = (const float*)d_in[25];
    const float* l2_w = (const float*)d_in[26]; const float* l2_b = (const float*)d_in[27];
    const float* hg = (const float*)d_in[28]; const float* hb = (const float*)d_in[29];
    const float* hw = (const float*)d_in[30];
    const int*   topk = (const int*)d_in[31];
    float* out = (float*)d_out;
    (void)in_sizes; (void)n_in; (void)out_size;

    // Workspace layout (floats), all buffers row-padded for 64-row GEMM tiles.
    float* w = (float*)d_ws;
    float* xb   = w; w += (size_t)NR * DD;        // posenc output (x before pre-LN)
    float* xb2  = w; w += (size_t)NR * DD;        // LN output / h buffer
    float* x3   = w; w += (size_t)NR * DD;        // running x state
    float* ob   = w; w += (size_t)NR * DD;        // attn-out / corr tmp
    float* qb   = w; w += (size_t)NR * QKD;
    float* kb   = w; w += (size_t)NR * QKD;
    float* qkvb = w; w += (size_t)NR * TD;
    float* gb   = w; w += (size_t)NR * FFD;
    float* tb   = w; w += (size_t)(DD + 16) * NP; // transposed feat / V (16-row slack)
    float* sb   = w; w += (size_t)NR * NP;        // attention scores/probs
    size_t need = (size_t)(w - (float*)d_ws) * sizeof(float);
    if (ws_size < need) return;

    dim3 blk128(128), blk256(256);
    auto gemm = [&](const float* A, int lda, const float* Wm, int ldb,
                    const float* bias, float* C, int ldc,
                    int M, int Nc, int K, float alpha, int act, int accum) {
        dim3 grid((Nc + 63) / 64, (M + 63) / 64);
        gemm_bf16_wmma<<<grid, blk128, 0, stream>>>(A, lda, Wm, ldb, bias, C, ldc,
                                                    M, Nc, K, alpha, act, accum);
    };

    const float inv16  = 1.0f / 16.0f;                 // 1/sqrt(QKD)
    const float invs48 = 0.14433756729740643f;         // 1/sqrt(48)

    for (int b = 0; b < 2; b++) {
        const float* xin = x_all + (size_t)b * SS * DD;

        // Positional encoding + cls, then pre-LN
        posenc_kernel<<<(SP * DD + 255) / 256, blk256, 0, stream>>>(
            xin, k7w, k7b, k5w, k5b, k3w, k3b, xb);
        copy_kernel<<<(DD + 255) / 256, blk256, 0, stream>>>(cls, xb, DD);
        ln_kernel<<<NN, blk256, 0, stream>>>(xb, nullptr, pre_g, pre_b, xb2, DD, 0, DD);

        // Sparse correlation attention: q, k projections; sim = q k^T / 16
        gemm(xb2 + DD, DD, q_w, DD, q_b, qb, QKD, SP, QKD, DD, 1.f, 0, 0);
        gemm(xb2 + DD, DD, k_w, DD, k_b, kb, QKD, SP, QKD, DD, 1.f, 0, 0);
        gemm(qb, QKD, kb, QKD, nullptr, sb, NP, SP, NP, QKD, inv16, 0, 0);
        softmax_topk_kernel<<<SP, blk256, 0, stream>>>(sb, NP, SP, topk);
        transpose_kernel<<<dim3((NP + 255) / 256, DD), blk256, 0, stream>>>(
            xb2 + DD, DD, SP, tb, NP);
        gemm(sb, NP, tb, NP, nullptr, ob, DD, SP, DD, NP, 1.f, 0, 0);
        ln_kernel<<<SP, blk256, 0, stream>>>(xb2 + DD, ob, corr_g, corr_b, x3 + DD, DD, DD, DD);
        copy_kernel<<<(DD + 255) / 256, blk256, 0, stream>>>(xb2, x3, DD); // cls row unchanged

        // Transformer blocks (shared weights, DEPTH=2)
        for (int dth = 0; dth < 2; dth++) {
            ln_kernel<<<NN, blk256, 0, stream>>>(x3, nullptr, n1_g, n1_b, xb2, DD, 0, DD);
            gemm(xb2, DD, qkv_w, DD, qkv_b, qkvb, TD, NN, TD, DD, 1.f, 0, 0);
            transpose_kernel<<<dim3((NP + 255) / 256, DD), blk256, 0, stream>>>(
                qkvb + 2 * DD, TD, NN, tb, NP);                    // V^T (768 x NP)
            for (int h = 0; h < NH; h++) {
                gemm(qkvb + h * HDIM, TD, qkvb + DD + h * HDIM, TD, nullptr,
                     sb, NP, NN, NP, HDIM, invs48, 0, 0);          // scores
                softmax_topk_kernel<<<NN, blk256, 0, stream>>>(sb, NP, NN, nullptr);
                gemm(sb, NP, tb + (size_t)h * HDIM * NP, NP, nullptr,
                     ob + h * HDIM, DD, NN, HDIM, NP, 1.f, 0, 0);  // probs @ V
            }
            gemm(ob, DD, out_w, DD, out_b, x3, DD, NN, DD, DD, 1.f, 0, 1);   // x += o W_o^T + b
            ln_kernel<<<NN, blk256, 0, stream>>>(x3, nullptr, n2_g, n2_b, xb2, DD, 0, DD);
            gemm(xb2, DD, l1_w, DD, l1_b, gb, FFD, NN, FFD, DD, 1.f, 1, 0);  // gelu(h W1^T+b1)
            gemm(gb, FFD, l2_w, FFD, l2_b, x3, DD, NN, DD, FFD, 1.f, 0, 1);  // x += g W2^T+b2
        }
        head_kernel<<<1, blk256, 0, stream>>>(x3, hg, hb, hw, out + b);
    }
}
